// VRWKV_SpatialMix_54769422958607
// MI455X (gfx1250) — matrix operand order repaired and checked by reading, compiled, not run on hardware
//
#include <hip/hip_runtime.h>
#include <stdint.h>

// Problem constants (match reference)
#define CDIM   256
#define BDIM   8
#define TDIM   4096
#define BTDIM  (BDIM * TDIM)      // 32768
#define NCHUNK 32
#define CHUNKL (TDIM / NCHUNK)    // 128

typedef __attribute__((ext_vector_type(16))) __bf16 bf16x16;
typedef __attribute__((ext_vector_type(8)))  float  f32x8;

union Frag {
    bf16x16  v;
    uint32_t u[8];
};

__device__ __forceinline__ uint16_t f2bf(float f) {
    uint32_t u = __float_as_uint(f);
    uint32_t r = (u + 0x7FFFu + ((u >> 16) & 1u)) >> 16;   // round-nearest-even
    return (uint16_t)r;
}

// A fragment (16x32 bf16, ISA layout): per lane two contiguous 16B loads
__device__ __forceinline__ bf16x16 load_fragA(const uint16_t* p) {
    uint4 t0 = *(const uint4*)(p);        // K = h*8 + 0..7
    uint4 t1 = *(const uint4*)(p + 16);   // K = h*8 + 16..23
    Frag F;
    F.u[0] = t0.x; F.u[1] = t0.y; F.u[2] = t0.z; F.u[3] = t0.w;
    F.u[4] = t1.x; F.u[5] = t1.y; F.u[6] = t1.z; F.u[7] = t1.w;
    return F.v;
}

// B fragment (32x16 bf16, ISA layout): per lane two contiguous 16B loads
__device__ __forceinline__ bf16x16 load_fragB(const uint16_t* p) {
    uint4 t0 = *(const uint4*)(p);        // K = h*8 + 0..7
    uint4 t1 = *(const uint4*)(p + 16);   // K = h*8 + 16..23
    Frag F;
    F.u[0] = t0.x; F.u[1] = t0.y; F.u[2] = t0.z; F.u[3] = t0.w;
    F.u[4] = t1.x; F.u[5] = t1.y; F.u[6] = t1.z; F.u[7] = t1.w;
    return F.v;
}

__device__ __forceinline__ f32x8 wmma_bf16(bf16x16 A, bf16x16 B, f32x8 C) {
    return __builtin_amdgcn_wmma_f32_16x16x32_bf16(false, A, false, B,
                                                   (short)0, C, false, false);
}

// ---------------------------------------------------------------------------
// f32 -> packed bf16 conversion (2 elems / thread)
// ---------------------------------------------------------------------------
__global__ __launch_bounds__(256)
void cvt_f32_bf16(const float* __restrict__ in, uint32_t* __restrict__ out, int n2) {
    int i = blockIdx.x * 256 + threadIdx.x;
    if (i < n2) {
        float a = in[2 * i + 0];
        float b = in[2 * i + 1];
        out[i] = (uint32_t)f2bf(a) | ((uint32_t)f2bf(b) << 16);
    }
}

// ---------------------------------------------------------------------------
// Fused k/v/r projection, 2x2 register-blocked:
// each wave computes a 32x32 output tile for all three matrices
// (12 WMMAs per K-step, A fragments shared across 3 weights x 2 N-subtiles).
// out[m,d] = sum_c x[m,c] * W[d,c]     (einsum 'btc,dc->btd')
// grid: (BT/32)*(C/32) = 1024*8 = 8192 waves = 1024 blocks of 8 waves.
// ---------------------------------------------------------------------------
__global__ __launch_bounds__(256)
void gemm3_wmma(const uint16_t* __restrict__ xb,
                const uint16_t* __restrict__ wk,
                const uint16_t* __restrict__ wv,
                const uint16_t* __restrict__ wr,
                float* __restrict__ kout,
                float* __restrict__ vout,
                float* __restrict__ rout) {
    const int wave = threadIdx.x >> 5;
    const int lane = threadIdx.x & 31;
    const int h    = lane >> 4;
    const int li   = lane & 15;
    const int tile = blockIdx.x * 8 + wave;
    const int mg   = tile >> 3;          // 0..1023 : 32-row group
    const int ng   = tile & 7;           // 0..7    : 32-col group

    const uint16_t* a0 = xb + (size_t)(mg * 32 + li) * CDIM + h * 8;
    const uint16_t* a1 = a0 + (size_t)16 * CDIM;

    const uint16_t* Wm[3] = {wk, wv, wr};
    const uint16_t* bp[3][2];
#pragma unroll
    for (int m = 0; m < 3; ++m) {
        bp[m][0] = Wm[m] + (size_t)(ng * 32 + li) * CDIM + h * 8;
        bp[m][1] = bp[m][0] + (size_t)16 * CDIM;
    }

    const f32x8 zero = {0.f, 0.f, 0.f, 0.f, 0.f, 0.f, 0.f, 0.f};
    f32x8 acc[3][2][2];
#pragma unroll
    for (int m = 0; m < 3; ++m)
#pragma unroll
        for (int mi = 0; mi < 2; ++mi)
#pragma unroll
            for (int ni = 0; ni < 2; ++ni) acc[m][mi][ni] = zero;

#pragma unroll
    for (int kk = 0; kk < CDIM; kk += 32) {
        bf16x16 A0 = load_fragA(a0 + kk);
        bf16x16 A1 = load_fragA(a1 + kk);
#pragma unroll
        for (int m = 0; m < 3; ++m) {
            bf16x16 B0 = load_fragB(bp[m][0] + kk);
            bf16x16 B1 = load_fragB(bp[m][1] + kk);
            acc[m][0][0] = wmma_bf16(A0, B0, acc[m][0][0]);
            acc[m][1][0] = wmma_bf16(A1, B0, acc[m][1][0]);
            acc[m][0][1] = wmma_bf16(A0, B1, acc[m][0][1]);
            acc[m][1][1] = wmma_bf16(A1, B1, acc[m][1][1]);
        }
    }

    float* Om[3] = {kout, vout, rout};
#pragma unroll
    for (int m = 0; m < 3; ++m)
#pragma unroll
        for (int mi = 0; mi < 2; ++mi)
#pragma unroll
            for (int ni = 0; ni < 2; ++ni)
#pragma unroll
                for (int g = 0; g < 8; ++g) {
                    size_t idx = (size_t)(mg * 32 + mi * 16 + h * 8 + g) * CDIM
                               + ng * 32 + ni * 16 + li;
                    Om[m][idx] = acc[m][mi][ni][g];
                }
}

// ---------------------------------------------------------------------------
// Output projection GEMM, same 2x2 blocking: out[m,d] = sum_c z[m,c]*Wo[d,c]
// ---------------------------------------------------------------------------
__global__ __launch_bounds__(256)
void gemm1_wmma(const uint16_t* __restrict__ zb,
                const uint16_t* __restrict__ wo,
                float* __restrict__ out) {
    const int wave = threadIdx.x >> 5;
    const int lane = threadIdx.x & 31;
    const int h    = lane >> 4;
    const int li   = lane & 15;
    const int tile = blockIdx.x * 8 + wave;
    const int mg   = tile >> 3;
    const int ng   = tile & 7;

    const uint16_t* a0 = zb + (size_t)(mg * 32 + li) * CDIM + h * 8;
    const uint16_t* a1 = a0 + (size_t)16 * CDIM;
    const uint16_t* b0 = wo + (size_t)(ng * 32 + li) * CDIM + h * 8;
    const uint16_t* b1 = b0 + (size_t)16 * CDIM;

    const f32x8 zero = {0.f, 0.f, 0.f, 0.f, 0.f, 0.f, 0.f, 0.f};
    f32x8 acc[2][2] = {{zero, zero}, {zero, zero}};

#pragma unroll
    for (int kk = 0; kk < CDIM; kk += 32) {
        bf16x16 A0 = load_fragA(a0 + kk);
        bf16x16 A1 = load_fragA(a1 + kk);
        bf16x16 B0 = load_fragB(b0 + kk);
        bf16x16 B1 = load_fragB(b1 + kk);
        acc[0][0] = wmma_bf16(A0, B0, acc[0][0]);
        acc[1][0] = wmma_bf16(A1, B0, acc[1][0]);
        acc[0][1] = wmma_bf16(A0, B1, acc[0][1]);
        acc[1][1] = wmma_bf16(A1, B1, acc[1][1]);
    }

#pragma unroll
    for (int mi = 0; mi < 2; ++mi)
#pragma unroll
        for (int ni = 0; ni < 2; ++ni)
#pragma unroll
            for (int g = 0; g < 8; ++g) {
                out[(size_t)(mg * 32 + mi * 16 + h * 8 + g) * CDIM
                    + ng * 32 + ni * 16 + li] = acc[mi][ni][g];
            }
}

// ---------------------------------------------------------------------------
// WKV chunked parallel scan.
// State (p,q,o): P = p*e^o, Q = q*e^o; update P' = e^w P + e^k v.
// Pass 1: per-(b,chunk,c) local aggregate from identity init.
// ---------------------------------------------------------------------------
__global__ __launch_bounds__(256)
void wkv_pass1(const float* __restrict__ kb, const float* __restrict__ vb,
               const float* __restrict__ decay,
               float* __restrict__ pa, float* __restrict__ qa, float* __restrict__ oa) {
    const int c = threadIdx.x;
    const int j = blockIdx.x;    // chunk
    const int b = blockIdx.y;    // batch
    const float w = decay[c] * (1.0f / TDIM);

    const size_t base = ((size_t)b * TDIM + (size_t)j * CHUNKL) * CDIM + c;
    const float* kp = kb + base;
    const float* vp = vb + base;

    float p = 0.f, q = 0.f, o = -1e38f;
    for (int i = 0; i < CHUNKL; ++i) {
        if (i + 8 < CHUNKL) {                       // global_prefetch_b8 path
            __builtin_prefetch(kp + (size_t)(i + 8) * CDIM, 0, 1);
            __builtin_prefetch(vp + (size_t)(i + 8) * CDIM, 0, 1);
        }
        float kt = kp[(size_t)i * CDIM];
        float vt = vp[(size_t)i * CDIM];
        float no = fmaxf(w + o, kt);
        float a2 = __expf(w + o - no);
        float b2 = __expf(kt - no);
        p = a2 * p + b2 * vt;
        q = a2 * q + b2;
        o = no;
    }
    const int idx = (b * NCHUNK + j) * CDIM + c;
    pa[idx] = p; qa[idx] = q; oa[idx] = o;
}

// Middle: exclusive scan over the 32 chunk aggregates per (b,c).
__global__ __launch_bounds__(256)
void wkv_scan_chunks(const float* __restrict__ decay,
                     const float* __restrict__ pa, const float* __restrict__ qa,
                     const float* __restrict__ oa,
                     float* __restrict__ pp, float* __restrict__ qq,
                     float* __restrict__ oo) {
    const int c = threadIdx.x;
    const int b = blockIdx.x;
    const float w  = decay[c] * (1.0f / TDIM);
    const float Lw = w * (float)CHUNKL;

    float p = 0.f, q = 0.f, o = -1e38f;
    for (int j = 0; j < NCHUNK; ++j) {
        const int idx = (b * NCHUNK + j) * CDIM + c;
        pp[idx] = p; qq[idx] = q; oo[idx] = o;   // state entering chunk j
        float pl = pa[idx], ql = qa[idx], ol = oa[idx];
        float on = fmaxf(o + Lw, ol);
        float e0 = __expf(o + Lw - on);
        float e1 = __expf(ol - on);
        p = e0 * p + e1 * pl;
        q = e0 * q + e1 * ql;
        o = on;
    }
}

// ---------------------------------------------------------------------------
// Pass 2 fused with LayerNorm + sigmoid gate + bf16 conversion.
// A block (256 threads = all C channels of one (b, chunk)) holds the full
// y row for each timestep in registers, so LN is a per-step block reduction:
// wave shuffle reduce (32 lanes) + 8-partial LDS combine. y never touches HBM.
// ---------------------------------------------------------------------------
__global__ __launch_bounds__(256)
void wkv_pass2_ln(const float* __restrict__ kb, const float* __restrict__ vb,
                  const float* __restrict__ rb,
                  const float* __restrict__ decay, const float* __restrict__ first,
                  const float* __restrict__ lnw, const float* __restrict__ lnb,
                  const float* __restrict__ pp, const float* __restrict__ qq,
                  const float* __restrict__ oo,
                  uint16_t* __restrict__ zb) {
    __shared__ float sm1[8];
    __shared__ float sm2[8];
    const int c    = threadIdx.x;
    const int lane = c & 31;
    const int wv   = c >> 5;
    const int j = blockIdx.x;
    const int b = blockIdx.y;
    const float w  = decay[c] * (1.0f / TDIM);
    const float u  = first[c] * (1.0f / TDIM);
    const float gw = lnw[c];
    const float gb = lnb[c];

    const int sidx = (b * NCHUNK + j) * CDIM + c;
    float p = pp[sidx], q = qq[sidx], o = oo[sidx];

    const size_t base = ((size_t)b * TDIM + (size_t)j * CHUNKL) * CDIM + c;
    const float* kp = kb + base;
    const float* vp = vb + base;
    const float* rp = rb + base;
    uint16_t*    zp = zb + base;

    for (int i = 0; i < CHUNKL; ++i) {
        if (i + 8 < CHUNKL) {
            __builtin_prefetch(kp + (size_t)(i + 8) * CDIM, 0, 1);
            __builtin_prefetch(vp + (size_t)(i + 8) * CDIM, 0, 1);
            __builtin_prefetch(rp + (size_t)(i + 8) * CDIM, 0, 1);
        }
        float kt = kp[(size_t)i * CDIM];
        float vt = vp[(size_t)i * CDIM];
        // y with bonus u (uses incoming state)
        float no = fmaxf(o, u + kt);
        float aa = __expf(o - no);
        float bb = __expf(u + kt - no);
        float y  = (aa * p + bb * vt) / (aa * q + bb);
        // state update
        float no2 = fmaxf(w + o, kt);
        float a2  = __expf(w + o - no2);
        float b2  = __expf(kt - no2);
        p = a2 * p + b2 * vt;
        q = a2 * q + b2;
        o = no2;

        // block-wide mean / mean-square over the 256 channels of timestep t
        float s1 = y, s2 = y * y;
#pragma unroll
        for (int off = 16; off > 0; off >>= 1) {
            s1 += __shfl_xor(s1, off, 32);
            s2 += __shfl_xor(s2, off, 32);
        }
        __syncthreads();                   // previous iteration's readers done
        if (lane == 0) { sm1[wv] = s1; sm2[wv] = s2; }
        __syncthreads();
        float t1 = 0.f, t2 = 0.f;
#pragma unroll
        for (int e = 0; e < 8; ++e) { t1 += sm1[e]; t2 += sm2[e]; }
        float mu  = t1 * (1.0f / CDIM);
        float var = t2 * (1.0f / CDIM) - mu * mu;

        float z  = (y - mu) * rsqrtf(var + 1e-5f) * gw + gb;
        float rv = rp[(size_t)i * CDIM];
        float sr = 1.0f / (1.0f + __expf(-rv));
        zp[(size_t)i * CDIM] = f2bf(sr * z);
    }
}

// ---------------------------------------------------------------------------
// Launch
// ---------------------------------------------------------------------------
extern "C" void kernel_launch(void* const* d_in, const int* in_sizes, int n_in,
                              void* d_out, int out_size, void* d_ws, size_t ws_size,
                              hipStream_t stream) {
    const float* x     = (const float*)d_in[0];
    const float* Wk    = (const float*)d_in[1];
    const float* Wv    = (const float*)d_in[2];
    const float* Wr    = (const float*)d_in[3];
    const float* Wo    = (const float*)d_in[4];
    const float* decay = (const float*)d_in[5];
    const float* first = (const float*)d_in[6];
    const float* lnw   = (const float*)d_in[7];
    const float* lnb   = (const float*)d_in[8];
    float* out = (float*)d_out;

    // Workspace carve-up (~130 MB)
    char* ws = (char*)d_ws;
    uint16_t* xb  = (uint16_t*)ws; ws += (size_t)BTDIM * CDIM * 2;
    uint16_t* wkb = (uint16_t*)ws; ws += (size_t)CDIM * CDIM * 2;
    uint16_t* wvb = (uint16_t*)ws; ws += (size_t)CDIM * CDIM * 2;
    uint16_t* wrb = (uint16_t*)ws; ws += (size_t)CDIM * CDIM * 2;
    uint16_t* wob = (uint16_t*)ws; ws += (size_t)CDIM * CDIM * 2;
    float* kbuf = (float*)ws; ws += (size_t)BTDIM * CDIM * 4;
    float* vbuf = (float*)ws; ws += (size_t)BTDIM * CDIM * 4;
    float* rbuf = (float*)ws; ws += (size_t)BTDIM * CDIM * 4;
    uint16_t* zb = (uint16_t*)ws; ws += (size_t)BTDIM * CDIM * 2;
    const size_t aggN = (size_t)BDIM * NCHUNK * CDIM;
    float* pa = (float*)ws; ws += aggN * 4;
    float* qa = (float*)ws; ws += aggN * 4;
    float* oa = (float*)ws; ws += aggN * 4;
    float* pp = (float*)ws; ws += aggN * 4;
    float* qq = (float*)ws; ws += aggN * 4;
    float* oo = (float*)ws; ws += aggN * 4;

    // 1) f32 -> bf16 conversions
    cvt_f32_bf16<<<(BTDIM * CDIM / 2) / 256, 256, 0, stream>>>(x, (uint32_t*)xb,
                                                              BTDIM * CDIM / 2);
    cvt_f32_bf16<<<(CDIM * CDIM / 2) / 256, 256, 0, stream>>>(Wk, (uint32_t*)wkb,
                                                              CDIM * CDIM / 2);
    cvt_f32_bf16<<<(CDIM * CDIM / 2) / 256, 256, 0, stream>>>(Wv, (uint32_t*)wvb,
                                                              CDIM * CDIM / 2);
    cvt_f32_bf16<<<(CDIM * CDIM / 2) / 256, 256, 0, stream>>>(Wr, (uint32_t*)wrb,
                                                              CDIM * CDIM / 2);
    cvt_f32_bf16<<<(CDIM * CDIM / 2) / 256, 256, 0, stream>>>(Wo, (uint32_t*)wob,
                                                              CDIM * CDIM / 2);

    // 2) fused k/v/r projection (WMMA, 32x32 tiles per wave)
    gemm3_wmma<<<(BTDIM / 32) * (CDIM / 32) / 8, 256, 0, stream>>>(
        xb, wkb, wvb, wrb, kbuf, vbuf, rbuf);

    // 3) chunked WKV scan; pass 2 fuses LayerNorm + sigmoid + bf16
    wkv_pass1<<<dim3(NCHUNK, BDIM), 256, 0, stream>>>(kbuf, vbuf, decay, pa, qa, oa);
    wkv_scan_chunks<<<BDIM, 256, 0, stream>>>(decay, pa, qa, oa, pp, qq, oo);
    wkv_pass2_ln<<<dim3(NCHUNK, BDIM), 256, 0, stream>>>(
        kbuf, vbuf, rbuf, decay, first, lnw, lnb, pp, qq, oo, zb);

    // 4) output projection (WMMA)
    gemm1_wmma<<<(BTDIM / 32) * (CDIM / 32) / 8, 256, 0, stream>>>(zb, wob, out);
}
